// RuleMemory_83897891160367
// MI455X (gfx1250) — matrix operand order repaired and checked by reading, not compiled, not run on hardware
//
#include <hip/hip_runtime.h>
#include <hip/hip_fp16.h>

typedef __attribute__((ext_vector_type(16))) _Float16 v16h_t;
typedef __attribute__((ext_vector_type(8)))  _Float16 v8h_t;
typedef __attribute__((ext_vector_type(4)))  _Float16 v4h_t;
typedef __attribute__((ext_vector_type(8)))  float    v8f_t;
typedef __attribute__((ext_vector_type(4)))  float    v4f_t;

#define NTROWS 8192
#define UVDIM  4096
#define SDIM   128
#define RDIM   256
#define LOGEPS (-13.815510557964274f)   /* log(1e-6) */

// ---------------------------------------------------------------------------
// WMMA helpers (CDNA5 V_WMMA_F32_16X16X32_F16, wave32)
// ---------------------------------------------------------------------------
__device__ __forceinline__ v8f_t wmma16(v16h_t a, v16h_t b, v8f_t c) {
  return __builtin_amdgcn_wmma_f32_16x16x32_f16(false, a, false, b,
                                                (short)0, c, false, false);
}

// A fragment (16x32 f16): lane L holds row m=L&15, halves kh=L>>4.
__device__ __forceinline__ v16h_t load_a_frag(const _Float16* base, int ld,
                                              int k0, int lane) {
  int m = lane & 15, kh = (lane >> 4) & 1;
  const _Float16* p = base + (size_t)m * ld + k0 + 8 * kh;
  v8h_t lo = *(const v8h_t*)p;
  v8h_t hi = *(const v8h_t*)(p + 16);
  v16h_t r;
#pragma unroll
  for (int i = 0; i < 8; ++i) { r[i] = lo[i]; r[i + 8] = hi[i]; }
  return r;
}

// B fragment (32x16 f16): lane L holds column n=L&15, k contiguous per column.
__device__ __forceinline__ v16h_t load_b_frag(const _Float16* base, int ld,
                                              int n0, int k0, int lane) {
  int n = lane & 15, kh = (lane >> 4) & 1;
  const _Float16* p = base + (size_t)(n0 + n) * ld + k0 + 8 * kh;
  v8h_t lo = *(const v8h_t*)p;
  v8h_t hi = *(const v8h_t*)(p + 16);
  v16h_t r;
#pragma unroll
  for (int i = 0; i < 8; ++i) { r[i] = lo[i]; r[i + 8] = hi[i]; }
  return r;
}

// ---------------------------------------------------------------------------
// K0: priors / validity mask (single block)
// ---------------------------------------------------------------------------
__global__ void k_priors(const float* __restrict__ write_mass,
                         const float* __restrict__ ema_conf,
                         float* __restrict__ prior,
                         float* __restrict__ conf_prior) {
  __shared__ float red[256];
  int tid = threadIdx.x;
  float m1 = 0.f, m2 = 0.f;
  for (int i = tid; i < UVDIM; i += 256) {
    m1 = fmaxf(m1, log1pf(write_mass[i]));
    m2 = fmaxf(m2, ema_conf[i]);
  }
  red[tid] = m1; __syncthreads();
  for (int s = 128; s > 0; s >>= 1) { if (tid < s) red[tid] = fmaxf(red[tid], red[tid + s]); __syncthreads(); }
  float M1 = red[0]; __syncthreads();
  red[tid] = m2; __syncthreads();
  for (int s = 128; s > 0; s >>= 1) { if (tid < s) red[tid] = fmaxf(red[tid], red[tid + s]); __syncthreads(); }
  float M2 = red[0]; __syncthreads();
  float inv1 = 1.0f / fmaxf(M1, 1.0f);
  float inv2 = 1.0f / fmaxf(M2, 1e-6f);
  for (int i = tid; i < UVDIM; i += 256) {
    float wm = write_mass[i];
    float cp = ema_conf[i] * inv2;
    conf_prior[i] = cp;
    float p = 0.5f * (log1pf(wm) * inv1) + 0.5f * cp;
    prior[i] = (wm > 0.f) ? p : -1e30f;   // sentinel => invalid cell
  }
}

// ---------------------------------------------------------------------------
// K1: proto prep. 4096 blocks x 128 threads.
// ---------------------------------------------------------------------------
__global__ void k_protos(const float* __restrict__ sig_proto,
                         const float* __restrict__ delta_proto,
                         _Float16* __restrict__ spn,
                         _Float16* __restrict__ sigT,
                         _Float16* __restrict__ deltaT) {
  __shared__ float red[128];
  int uv = blockIdx.x, t = threadIdx.x;
  float x = sig_proto[(size_t)uv * SDIM + t] + 1e-6f;
  red[t] = x * x; __syncthreads();
  for (int s = 64; s > 0; s >>= 1) { if (t < s) red[t] += red[t + s]; __syncthreads(); }
  float inv = 1.0f / fmaxf(sqrtf(red[0]), 1e-12f);
  spn[(size_t)uv * SDIM + t] = (_Float16)(x * inv);
  sigT[(size_t)t * UVDIM + uv] = (_Float16)sig_proto[(size_t)uv * SDIM + t];
  deltaT[(size_t)t * UVDIM + uv]         = (_Float16)delta_proto[(size_t)uv * RDIM + t];
  deltaT[(size_t)(t + 128) * UVDIM + uv] = (_Float16)delta_proto[(size_t)uv * RDIM + t + 128];
}

// ---------------------------------------------------------------------------
// K2: normalize q_sigma -> f16 A source. 8192 blocks x 128 threads.
// ---------------------------------------------------------------------------
__global__ void k_qsig(const float* __restrict__ q_sigma,
                       _Float16* __restrict__ qs) {
  __shared__ float red[128];
  int row = blockIdx.x, t = threadIdx.x;
  float x = q_sigma[(size_t)row * SDIM + t];
  red[t] = x * x; __syncthreads();
  for (int s = 64; s > 0; s >>= 1) { if (t < s) red[t] += red[t + s]; __syncthreads(); }
  float inv = 1.0f / fmaxf(sqrtf(red[0]), 1e-12f);
  qs[(size_t)row * SDIM + t] = (_Float16)(x * inv);
}

// ---------------------------------------------------------------------------
// K3: fused sig-GEMM + logits + masked softmax + argmax. 512 blocks x 256 thr.
// Dynamic LDS: logits[16][4096] f32 + prior[4096] + luT/lbT[64][16] + stats.
// B-fragments double-buffered across N-tiles; two independent accumulators.
// ---------------------------------------------------------------------------
__global__ void __launch_bounds__(256)
k_main(const float* __restrict__ q_u,
       const float* __restrict__ q_b,
       const _Float16* __restrict__ qs,
       const _Float16* __restrict__ spn,
       const float* __restrict__ prior,
       const float* __restrict__ conf_prior,
       float* __restrict__ out_weights,
       float* __restrict__ out_conf,
       float* __restrict__ out_topw,
       _Float16* __restrict__ w16) {
  extern __shared__ char smem_raw[];
  float* Slog   = (float*)smem_raw;           // [16][4096]
  float* Sprior = Slog + 16 * UVDIM;          // [4096]
  float* SluT   = Sprior + UVDIM;             // [64][16]  (u-major, m contiguous)
  float* SlbT   = SluT + 1024;                // [64][16]  (v-major, m contiguous)
  float* Smax   = SlbT + 1024;                // [16]
  float* Ssum   = Smax + 16;                  // [16]
  int*   Sidx   = (int*)(Ssum + 16);          // [16]

  int tid = threadIdx.x, lane = tid & 31, wv = tid >> 5;
  int row0 = blockIdx.x * 16;

  for (int i = tid; i < UVDIM; i += 256) Sprior[i] = prior[i];
  for (int i = tid; i < 1024; i += 256) {
    int m = i & 15, j = i >> 4;
    SluT[i] = logf(q_u[(size_t)(row0 + m) * 64 + j]);
    SlbT[i] = logf(q_b[(size_t)(row0 + m) * 64 + j]);
  }
  __syncthreads();

  // A fragments: 16 rows x S=128 (4 k-tiles of 32)
  const _Float16* Abase = qs + (size_t)row0 * SDIM;
  v16h_t a0 = load_a_frag(Abase, SDIM, 0,  lane);
  v16h_t a1 = load_a_frag(Abase, SDIM, 32, lane);
  v16h_t a2 = load_a_frag(Abase, SDIM, 64, lane);
  v16h_t a3 = load_a_frag(Abase, SDIM, 96, lane);

  int n = lane & 15, kh = (lane >> 4) & 1;

  // prologue: tile 0 B frags
  int colP = wv * 512;
  v16h_t b0 = load_b_frag(spn, SDIM, colP, 0,  lane);
  v16h_t b1 = load_b_frag(spn, SDIM, colP, 32, lane);
  v16h_t b2 = load_b_frag(spn, SDIM, colP, 64, lane);
  v16h_t b3 = load_b_frag(spn, SDIM, colP, 96, lane);

  for (int nt = 0; nt < 32; ++nt) {
    int col0 = wv * 512 + nt * 16;
    // issue next tile's loads before consuming current tile
    int colN = wv * 512 + ((nt + 1) & 31) * 16;
    v16h_t c0 = load_b_frag(spn, SDIM, colN, 0,  lane);
    v16h_t c1 = load_b_frag(spn, SDIM, colN, 32, lane);
    v16h_t c2 = load_b_frag(spn, SDIM, colN, 64, lane);
    v16h_t c3 = load_b_frag(spn, SDIM, colN, 96, lane);

    v8f_t accA = {}, accB = {};
    accA = wmma16(a0, b0, accA);
    accB = wmma16(a1, b1, accB);
    accA = wmma16(a2, b2, accA);
    accB = wmma16(a3, b3, accB);

    int col = col0 + n;
    int u = col0 >> 6;
    int v = (col0 & 63) + n;
    float pr = Sprior[col];
    bool inval = (pr < -1e20f);
    const v4f_t* LU = (const v4f_t*)(SluT + u * 16 + 8 * kh);
    const v4f_t* LB = (const v4f_t*)(SlbT + v * 16 + 8 * kh);
    v4f_t lu0 = LU[0], lu1 = LU[1];
    v4f_t lb0 = LB[0], lb1 = LB[1];
#pragma unroll
    for (int r = 0; r < 8; ++r) {
      float luv = (r < 4) ? lu0[r & 3] : lu1[r & 3];
      float lbv = (r < 4) ? lb0[r & 3] : lb1[r & 3];
      float jl = fmaxf(luv + lbv, LOGEPS);
      float lg = jl + pr + 0.5f + 0.5f * (accA[r] + accB[r]);
      Slog[(r + 8 * kh) * UVDIM + col] = inval ? -1e9f : lg;
    }
    b0 = c0; b1 = c1; b2 = c2; b3 = c3;
  }
  __syncthreads();

  // Softmax stats: 8 waves x 2 rows each; max+argmax then sum of exp (float4)
  for (int rr = 0; rr < 2; ++rr) {
    int m = wv * 2 + rr;
    const v4f_t* Lr = (const v4f_t*)(Slog + m * UVDIM);
    float mx = -3e38f; int mi = 0;
    for (int it = 0; it < 32; ++it) {
      int idx = it * 32 + lane;
      v4f_t x = Lr[idx];
#pragma unroll
      for (int c = 0; c < 4; ++c) {
        float xv = x[c]; int ii = idx * 4 + c;
        if (xv > mx) { mx = xv; mi = ii; }
      }
    }
    for (int off = 16; off > 0; off >>= 1) {
      float ox = __shfl_xor(mx, off, 32);
      int   oi = __shfl_xor(mi, off, 32);
      if (ox > mx || (ox == mx && oi < mi)) { mx = ox; mi = oi; }
    }
    float sum = 0.f;
    for (int it = 0; it < 32; ++it) {
      v4f_t x = Lr[it * 32 + lane];
      sum += expf(x[0] - mx) + expf(x[1] - mx) + expf(x[2] - mx) + expf(x[3] - mx);
    }
    for (int off = 16; off > 0; off >>= 1) sum += __shfl_xor(sum, off, 32);
    if (lane == 0) { Smax[m] = mx; Ssum[m] = fmaxf(sum, 1e-6f); Sidx[m] = mi; }
  }
  __syncthreads();

  // Emit weights (f32 to output, f16 to scratch), float4-vectorized
  for (int i = tid; i < 16 * UVDIM / 4; i += 256) {
    int m = i >> 10, c4 = i & 1023;
    v4f_t lg = ((const v4f_t*)Slog)[i];
    float mxm = Smax[m], inv = 1.0f / Ssum[m];
    v4f_t w; v4h_t wh;
#pragma unroll
    for (int c = 0; c < 4; ++c) {
      float wc = (lg[c] <= -5e8f) ? 0.f : expf(lg[c] - mxm) * inv;
      w[c] = wc; wh[c] = (_Float16)wc;
    }
    size_t g4 = (size_t)(row0 + m) * (UVDIM / 4) + c4;
    ((v4f_t*)out_weights)[g4] = w;
    ((v4h_t*)w16)[g4] = wh;
  }
  // Per-row scalars
  if (tid < 16) {
    int m = tid, ti = Sidx[m];
    float lg = Slog[m * UVDIM + ti];
    float topw = (lg <= -5e8f) ? 0.f : expf(lg - Smax[m]) / Ssum[m];
    int u = ti >> 6, v = ti & 63;
    float jl = fmaxf(SluT[u * 16 + m] + SlbT[v * 16 + m], LOGEPS);
    float sig = lg - jl - Sprior[ti];           // = 0.5*(1+s)
    float mc = fminf(fmaxf(topw * conf_prior[ti] * sig, 0.f), 1.f);
    out_topw[row0 + m] = topw;
    out_conf[row0 + m] = (lg <= -5e8f) ? 0.f : mc;
  }
}

// ---------------------------------------------------------------------------
// K4: output GEMMs: [16 x 4096] w16  x  deltaT[256 cols] / sigT[128 cols]
// A tile staged in 128KB LDS via ASYNC global->LDS copy; A/B double-buffered.
// ---------------------------------------------------------------------------
__global__ void __launch_bounds__(256)
k_outgemm(const _Float16* __restrict__ w16,
          const _Float16* __restrict__ deltaT,
          const _Float16* __restrict__ sigT,
          float* __restrict__ out_delta,
          float* __restrict__ out_sig) {
  extern __shared__ char smem_raw2[];
  _Float16* Sw = (_Float16*)smem_raw2;            // [16][4096], LDS offset 0
  int tid = threadIdx.x, lane = tid & 31, wv = tid >> 5;
  int row0 = blockIdx.x * 16;

  // Async global->LDS tile copy (ASYNCcnt path, bypasses VGPRs).
  // Dynamic LDS is the only LDS segment here -> raw LDS offset == i*16.
  {
    const _Float16* gbase = w16 + (size_t)row0 * UVDIM;
    for (int i = tid; i < 16 * UVDIM / 8; i += 256) {
      const _Float16* gp = gbase + (size_t)i * 8;
      unsigned loff = (unsigned)i * 16u;
      asm volatile("global_load_async_to_lds_b128 %0, %1, off"
                   :: "v"(loff), "v"(gp) : "memory");
    }
    asm volatile("s_wait_asynccnt 0x0" ::: "memory");
  }
  __syncthreads();

  // per-wave B sources (wave-uniform)
  const _Float16* Bj[3]; int n0j[3]; float* Oj[3]; int ncj[3];
#pragma unroll
  for (int j = 0; j < 3; ++j) {
    int gt = wv * 3 + j;
    Bj[j] = (gt < 16) ? deltaT : sigT;
    n0j[j] = (gt < 16) ? gt * 16 : (gt - 16) * 16;
    Oj[j] = (gt < 16) ? out_delta : out_sig;
    ncj[j] = (gt < 16) ? RDIM : SDIM;
  }

  // prologue
  v16h_t a  = load_a_frag(Sw, UVDIM, 0, lane);
  v16h_t b0 = load_b_frag(Bj[0], UVDIM, n0j[0], 0, lane);
  v16h_t b1 = load_b_frag(Bj[1], UVDIM, n0j[1], 0, lane);
  v16h_t b2 = load_b_frag(Bj[2], UVDIM, n0j[2], 0, lane);

  v8f_t acc0 = {}, acc1 = {}, acc2 = {};
  for (int kt = 0; kt < 128; ++kt) {
    int kn = ((kt + 1) & 127) * 32;
    v16h_t an = load_a_frag(Sw, UVDIM, kn, lane);
    v16h_t c0 = load_b_frag(Bj[0], UVDIM, n0j[0], kn, lane);
    v16h_t c1 = load_b_frag(Bj[1], UVDIM, n0j[1], kn, lane);
    v16h_t c2 = load_b_frag(Bj[2], UVDIM, n0j[2], kn, lane);

    acc0 = wmma16(a, b0, acc0);
    acc1 = wmma16(a, b1, acc1);
    acc2 = wmma16(a, b2, acc2);

    a = an; b0 = c0; b1 = c1; b2 = c2;
  }

  int n = lane & 15, kh = (lane >> 4) & 1;
  v8f_t accs[3] = {acc0, acc1, acc2};
#pragma unroll
  for (int j = 0; j < 3; ++j) {
    int c = n0j[j] + n;
#pragma unroll
    for (int r = 0; r < 8; ++r) {
      int m = r + 8 * kh;
      Oj[j][(size_t)(row0 + m) * ncj[j] + c] = accs[j][r];
    }
  }
}

// ---------------------------------------------------------------------------
// Host launcher
// ---------------------------------------------------------------------------
extern "C" void kernel_launch(void* const* d_in, const int* in_sizes, int n_in,
                              void* d_out, int out_size, void* d_ws, size_t ws_size,
                              hipStream_t stream) {
  const float* q_u        = (const float*)d_in[0];   // [4,2048,64]
  const float* q_b        = (const float*)d_in[1];   // [4,2048,64]
  const float* q_sigma    = (const float*)d_in[2];   // [4,2048,128]
  const float* delta_prot = (const float*)d_in[3];   // [64,64,256]
  const float* sig_prot   = (const float*)d_in[4];   // [64,64,128]
  const float* write_mass = (const float*)d_in[5];   // [64,64]
  const float* ema_conf   = (const float*)d_in[6];   // [64,64]

  float* out = (float*)d_out;
  float* out_delta   = out;                                   // 8192*256
  float* out_sig     = out_delta + (size_t)NTROWS * RDIM;     // 8192*128
  float* out_conf    = out_sig + (size_t)NTROWS * SDIM;       // 8192
  float* out_weights = out_conf + NTROWS;                     // 8192*4096
  float* out_topw    = out_weights + (size_t)NTROWS * UVDIM;  // 8192

  char* ws = (char*)d_ws;
  size_t off = 0;
  float* prior      = (float*)(ws + off); off += (size_t)UVDIM * 4;
  float* conf_prior = (float*)(ws + off); off += (size_t)UVDIM * 4;
  _Float16* spn     = (_Float16*)(ws + off); off += (size_t)UVDIM * SDIM * 2;
  _Float16* sigT    = (_Float16*)(ws + off); off += (size_t)SDIM * UVDIM * 2;
  _Float16* deltaT  = (_Float16*)(ws + off); off += (size_t)RDIM * UVDIM * 2;
  _Float16* qs      = (_Float16*)(ws + off); off += (size_t)NTROWS * SDIM * 2;
  _Float16* w16     = (_Float16*)(ws + off); off += (size_t)NTROWS * UVDIM * 2;

  const int SMEM_MAIN = (16 * UVDIM + UVDIM + 1024 + 1024 + 48) * 4; // ~287 KB
  const int SMEM_OUT  = 16 * UVDIM * 2;                              // 128 KB
  hipFuncSetAttribute((const void*)k_main,    hipFuncAttributeMaxDynamicSharedMemorySize, SMEM_MAIN);
  hipFuncSetAttribute((const void*)k_outgemm, hipFuncAttributeMaxDynamicSharedMemorySize, SMEM_OUT);

  k_priors<<<1, 256, 0, stream>>>(write_mass, ema_conf, prior, conf_prior);
  k_protos<<<UVDIM, 128, 0, stream>>>(sig_prot, delta_prot, spn, sigT, deltaT);
  k_qsig<<<NTROWS, 128, 0, stream>>>(q_sigma, qs);
  k_main<<<NTROWS / 16, 256, SMEM_MAIN, stream>>>(q_u, q_b, qs, spn, prior, conf_prior,
                                                  out_weights, out_conf, out_topw, w16);
  k_outgemm<<<NTROWS / 16, 256, SMEM_OUT, stream>>>(w16, deltaT, sigT, out_delta, out_sig);
}
